// SpkerClassifier_72490458022262
// MI455X (gfx1250) — compile-verified, matrix operand order checked by reading, not run on hardware
//
#include <hip/hip_runtime.h>
#include <hip/hip_bf16.h>

typedef __attribute__((ext_vector_type(16))) __bf16 v16bf;
typedef __attribute__((ext_vector_type(8)))  __bf16 bf16x8;
typedef __attribute__((ext_vector_type(8)))  float  v8f;
typedef __attribute__((ext_vector_type(2)))  float  v2f;

// exact pointer element type expected by __builtin_amdgcn_global_load_async_to_lds_b128
typedef int v4i __attribute__((vector_size(16)));
typedef __attribute__((address_space(1))) v4i gv4i;   // global (AS1)
typedef __attribute__((address_space(3))) v4i lv4i;   // LDS (AS3)

constexpr float kSCALE   = 30.0f;
constexpr float kMARGIN  = 0.4f;
constexpr float kPENALTY = 0.1f;
constexpr int   kN       = 4096;
constexpr int   kD       = 768;
constexpr int   kL       = 20000;
constexpr int   kLPAD    = 20096;   // 157 * 128

#if defined(__has_builtin)
#if __has_builtin(__builtin_amdgcn_wmma_f32_16x16x4_f32)
#define USE_WMMA_F32 1
#endif
#if __has_builtin(__builtin_amdgcn_global_load_async_to_lds_b128) && \
    __has_builtin(__builtin_amdgcn_s_wait_asynccnt)
#define HAVE_ASYNC 1
#endif
#endif
#ifndef USE_WMMA_F32
#define USE_WMMA_F32 0
#endif
#ifndef HAVE_ASYNC
#define HAVE_ASYNC 0
#endif

union V8u  { v8f  v; float  f[8]; };
union V16u { v16bf v; bf16x8 h[2]; };

#if HAVE_ASYNC
__device__ __forceinline__ void async_copy16(const void* g, void* s) {
  __builtin_amdgcn_global_load_async_to_lds_b128((gv4i*)(void*)g, (lv4i*)(void*)s, 0, 0);
}
#endif

__device__ __forceinline__ void async_wait_then_barrier() {
#if HAVE_ASYNC
  __builtin_amdgcn_s_wait_asynccnt(0);
#endif
  __syncthreads();
}

// ---------------- Kernel 1: h = hidden @ clf_w^T + b  (fp32 WMMA) ----------------
#define H_LS 36   // fp32 LDS row stride (floats); 36*4=144B, 16B aligned

__global__ void __launch_bounds__(256)
linear_kernel(const float* __restrict__ hidden, const float* __restrict__ clfw,
              const float* __restrict__ clfb, float* __restrict__ hbuf) {
  __shared__ float hA[128 * H_LS];
  __shared__ float hB[128 * H_LS];
  const int t = threadIdx.x, lane = t & 31, wid = t >> 5;
  const int waveM = wid >> 2, waveN = wid & 3;
  const int rowBase = blockIdx.y * 128;
  const int colBase = blockIdx.x * 128;

  V8u acc[4][2];
  for (int i = 0; i < 4; ++i)
    for (int j = 0; j < 2; ++j)
      for (int e = 0; e < 8; ++e) acc[i][j].f[e] = 0.0f;

  const int ldr = t >> 1, half = t & 1;
  for (int k0 = 0; k0 < kD; k0 += 32) {
    {   // A tile: 128 rows x 32 K, row-major from hidden (verbatim copy)
      const float* g = hidden + (size_t)(rowBase + ldr) * kD + k0 + half * 16;
      float* s = &hA[ldr * H_LS + half * 16];
#if HAVE_ASYNC
#pragma unroll
      for (int j = 0; j < 4; ++j) async_copy16(g + j * 4, s + j * 4);
#else
#pragma unroll
      for (int j = 0; j < 4; ++j) *(float4*)(s + j * 4) = *(const float4*)(g + j * 4);
#endif
    }
    {   // B tile: rows of clf_w are the contraction dim -> already [n][k] (verbatim copy)
      const float* g = clfw + (size_t)(colBase + ldr) * kD + k0 + half * 16;
      float* s = &hB[ldr * H_LS + half * 16];
#if HAVE_ASYNC
#pragma unroll
      for (int j = 0; j < 4; ++j) async_copy16(g + j * 4, s + j * 4);
#else
#pragma unroll
      for (int j = 0; j < 4; ++j) *(float4*)(s + j * 4) = *(const float4*)(g + j * 4);
#endif
    }
    async_wait_then_barrier();
#if USE_WMMA_F32
#pragma unroll
    for (int kk = 0; kk < 32; kk += 4) {
      const int ksel = kk + ((lane >> 4) << 1);   // lanes 0-15: K 0,1  lanes 16-31: K 2,3
      v2f afrag[4], bfrag[2];
#pragma unroll
      for (int mt = 0; mt < 4; ++mt) {
        int m = waveM * 64 + mt * 16 + (lane & 15);
        afrag[mt] = *(const v2f*)&hA[m * H_LS + ksel];
      }
#pragma unroll
      for (int nt = 0; nt < 2; ++nt) {
        int n = waveN * 32 + nt * 16 + (lane & 15);
        bfrag[nt] = *(const v2f*)&hB[n * H_LS + ksel];
      }
#pragma unroll
      for (int mt = 0; mt < 4; ++mt)
#pragma unroll
        for (int nt = 0; nt < 2; ++nt)
          acc[mt][nt].v = __builtin_amdgcn_wmma_f32_16x16x4_f32(
              false, afrag[mt], false, bfrag[nt], (short)0, acc[mt][nt].v, false, false);
    }
#else
    for (int kk = 0; kk < 32; ++kk) {
#pragma unroll
      for (int mt = 0; mt < 4; ++mt)
#pragma unroll
        for (int nt = 0; nt < 2; ++nt) {
          int n = waveN * 32 + nt * 16 + (lane & 15);
          float bv = hB[n * H_LS + kk];
#pragma unroll
          for (int vi = 0; vi < 8; ++vi) {
            int m = waveM * 64 + mt * 16 + ((lane >> 4) << 3) + vi;
            acc[mt][nt].f[vi] += hA[m * H_LS + kk] * bv;
          }
        }
    }
#endif
    __syncthreads();
  }
#pragma unroll
  for (int mt = 0; mt < 4; ++mt)
    for (int nt = 0; nt < 2; ++nt)
      for (int vi = 0; vi < 8; ++vi) {
        int r = rowBase + waveM * 64 + mt * 16 + ((lane >> 4) << 3) + vi;
        int c = colBase + waveN * 32 + nt * 16 + (lane & 15);
        hbuf[(size_t)r * kD + c] = acc[mt][nt].f[vi] + clfb[c];
      }
}

// ---------------- Kernel 2: column-normalize weight -> bf16 B [kD x kLPAD] ----------------
__global__ void __launch_bounds__(256)
colnorm_kernel(const float* __restrict__ weight, __bf16* __restrict__ bbuf) {
  int l = blockIdx.x * 256 + threadIdx.x;
  if (l >= kLPAD) return;
  if (l < kL) {
    float s = 0.0f;
    for (int k = 0; k < kD; ++k) { float v = weight[(size_t)k * kL + l]; s += v * v; }
    float inv = rsqrtf(s);
    for (int k = 0; k < kD; ++k)
      bbuf[(size_t)k * kLPAD + l] = (__bf16)(weight[(size_t)k * kL + l] * inv);
  } else {
    for (int k = 0; k < kD; ++k) bbuf[(size_t)k * kLPAD + l] = (__bf16)0.0f;
  }
}

// ---------------- Kernel 3: row-normalize h -> bf16 A [kN x kD] ----------------
__global__ void __launch_bounds__(256)
rownorm_kernel(const float* __restrict__ hbuf, __bf16* __restrict__ abuf) {
  __shared__ float red[256];
  int row = blockIdx.x, t = threadIdx.x;
  const float* rp = hbuf + (size_t)row * kD;
  float s = 0.0f;
  for (int i = t; i < kD; i += 256) { float v = rp[i]; s += v * v; }
  red[t] = s;
  __syncthreads();
  for (int st = 128; st > 0; st >>= 1) {
    if (t < st) red[t] += red[t + st];
    __syncthreads();
  }
  float inv = rsqrtf(red[0]);
  for (int i = t; i < kD; i += 256) abuf[(size_t)row * kD + i] = (__bf16)(rp[i] * inv);
}

// ---- Kernel 4: cos = A @ B  (bf16 WMMA, fp32 accumulate, double-buffered LDS) ----
#define G_LS 40   // bf16 LDS row stride; 40*2=80B, 16B aligned at all fragment offsets
#define KT   (kD / 32)   // 24 K-tiles

__global__ void __launch_bounds__(256)
cos_gemm_kernel(const __bf16* __restrict__ A, const __bf16* __restrict__ B,
                float* __restrict__ out) {
  __shared__ __bf16 lA[2][128 * G_LS];
  __shared__ __bf16 lB[2][128 * G_LS];   // transposed: lB[buf][n][k]
  const int t = threadIdx.x, lane = t & 31, wid = t >> 5;
  const int waveM = wid >> 2, waveN = wid & 3;
  const int rowBase = blockIdx.y * 128;
  const int colBase = blockIdx.x * 128;

  V8u acc[4][2];
  for (int i = 0; i < 4; ++i)
    for (int j = 0; j < 2; ++j)
      for (int e = 0; e < 8; ++e) acc[i][j].f[e] = 0.0f;

  const int ldr = t >> 1, half = t & 1;   // A loader: 2 threads per row
  const int kB = t >> 3, segB = t & 7;    // B loader: 8 threads per K-row

  auto load_tile = [&](int k0, int buf) {
    {   // A tile 128x32 bf16, verbatim row-major copy -> async-eligible
      const __bf16* g = A + (size_t)(rowBase + ldr) * kD + k0 + half * 16;
      __bf16* s = &lA[buf][ldr * G_LS + half * 16];
#if HAVE_ASYNC
      async_copy16(g, s);
      async_copy16(g + 8, s + 8);
#else
      *(bf16x8*)(s)     = *(const bf16x8*)(g);
      *(bf16x8*)(s + 8) = *(const bf16x8*)(g + 8);
#endif
    }
    {   // B tile 32x128 bf16, transpose into lB[buf][n][k] (must go through VGPRs)
      const __bf16* g = B + (size_t)(k0 + kB) * kLPAD + colBase + segB * 16;
      bf16x8 b0 = *(const bf16x8*)(g);
      bf16x8 b1 = *(const bf16x8*)(g + 8);
#pragma unroll
      for (int i = 0; i < 8; ++i) {
        lB[buf][(segB * 16 + i)     * G_LS + kB] = b0[i];
        lB[buf][(segB * 16 + 8 + i) * G_LS + kB] = b1[i];
      }
    }
    if (k0 + 32 < kD)   // global_prefetch_b8 a tile ahead for the B stream
      __builtin_prefetch(B + (size_t)(k0 + 32 + kB) * kLPAD + colBase, 0, 1);
  };

  load_tile(0, 0);
  async_wait_then_barrier();

  for (int kt = 0; kt < KT; ++kt) {
    const int cur = kt & 1;
    if (kt + 1 < KT) load_tile((kt + 1) * 32, cur ^ 1);   // overlap next tile with compute

    const int khalf = (lane >> 4) << 3;   // lanes 0-15: K 0-7/16-23, lanes 16-31: K 8-15/24-31
    V16u afrag[4], bfrag[2];
#pragma unroll
    for (int mt = 0; mt < 4; ++mt) {
      int m = waveM * 64 + mt * 16 + (lane & 15);
      afrag[mt].h[0] = *(const bf16x8*)&lA[cur][m * G_LS + khalf];
      afrag[mt].h[1] = *(const bf16x8*)&lA[cur][m * G_LS + 16 + khalf];
    }
#pragma unroll
    for (int nt = 0; nt < 2; ++nt) {
      int n = waveN * 32 + nt * 16 + (lane & 15);
      bfrag[nt].h[0] = *(const bf16x8*)&lB[cur][n * G_LS + khalf];
      bfrag[nt].h[1] = *(const bf16x8*)&lB[cur][n * G_LS + 16 + khalf];
    }
#pragma unroll
    for (int mt = 0; mt < 4; ++mt)
#pragma unroll
      for (int nt = 0; nt < 2; ++nt)
        acc[mt][nt].v = __builtin_amdgcn_wmma_f32_16x16x32_bf16(
            false, afrag[mt].v, false, bfrag[nt].v, (short)0, acc[mt][nt].v, false, false);

    async_wait_then_barrier();   // next tile landed; prev buffer free for reuse
  }
#pragma unroll
  for (int mt = 0; mt < 4; ++mt)
    for (int nt = 0; nt < 2; ++nt)
      for (int vi = 0; vi < 8; ++vi) {
        int r = rowBase + waveM * 64 + mt * 16 + ((lane >> 4) << 3) + vi;
        int c = colBase + waveN * 32 + nt * 16 + (lane & 15);
        if (c < kL) out[(size_t)r * kL + c] = acc[mt][nt].f[vi];
      }
}

// ---------------- Kernel 5: per-row 6th-largest threshold ----------------
__device__ inline void topk_insert(float* b, float v) {
  if (v > b[5]) {
    b[5] = v;
#pragma unroll
    for (int i = 5; i > 0; --i)
      if (b[i] > b[i - 1]) { float tmp = b[i]; b[i] = b[i - 1]; b[i - 1] = tmp; }
  }
}
__device__ inline void topk_merge(float* a, const float* b) {
  float out[6]; int i = 0, j = 0;
#pragma unroll
  for (int k = 0; k < 6; ++k) {
    float av = a[i], bv = b[j];
    if (av >= bv) { out[k] = av; ++i; } else { out[k] = bv; ++j; }
  }
#pragma unroll
  for (int k = 0; k < 6; ++k) a[k] = out[k];
}

__global__ void __launch_bounds__(256)
topk_kernel(const float* __restrict__ cosm, float* __restrict__ thr) {
  __shared__ float sm[256 * 6];
  int row = blockIdx.x, t = threadIdx.x;
  const float* rp = cosm + (size_t)row * kL;
  float best[6];
  for (int i = 0; i < 6; ++i) best[i] = -3.0e38f;
  for (int c = t; c < kL; c += 256) {
    if (c + 256 < kL) __builtin_prefetch(rp + c + 256, 0, 1);
    topk_insert(best, rp[c]);
  }
  for (int i = 0; i < 6; ++i) sm[t * 6 + i] = best[i];
  __syncthreads();
  for (int st = 128; st > 0; st >>= 1) {
    if (t < st) topk_merge(&sm[t * 6], &sm[(t + st) * 6]);
    __syncthreads();
  }
  if (t == 0) thr[row] = sm[5];
}

// ---------------- Kernel 6: in-place margin/penalty/scale ----------------
__global__ void __launch_bounds__(256)
finalize_kernel(float* __restrict__ out, const int* __restrict__ labels,
                const float* __restrict__ thr) {
  size_t base = ((size_t)blockIdx.x * 256 + threadIdx.x) * 4;
  int row = (int)(base / kL);
  int col0 = (int)(base % kL);   // kL % 4 == 0: all 4 in same row
  float4 v = *(const float4*)(out + base);
  int lab = labels[row];
  float th = thr[row];
  float r[4] = {v.x, v.y, v.z, v.w};
#pragma unroll
  for (int j = 0; j < 4; ++j) {
    int col = col0 + j;
    float c = r[j];
    float x = (col == lab) ? (c - kMARGIN) : ((c >= th) ? (c + kPENALTY) : c);
    r[j] = kSCALE * x;
  }
  float4 o = {r[0], r[1], r[2], r[3]};
  *(float4*)(out + base) = o;
}

extern "C" void kernel_launch(void* const* d_in, const int* in_sizes, int n_in,
                              void* d_out, int out_size, void* d_ws, size_t ws_size,
                              hipStream_t stream) {
  (void)in_sizes; (void)n_in; (void)out_size; (void)ws_size;
  const float* hidden = (const float*)d_in[0];
  const int*   labels = (const int*)d_in[1];
  const float* weight = (const float*)d_in[2];
  const float* clfw   = (const float*)d_in[3];
  const float* clfb   = (const float*)d_in[4];
  float* out = (float*)d_out;

  char* ws = (char*)d_ws;
  float*  hbuf = (float*)ws;                                                  // 12.58 MB
  __bf16* abuf = (__bf16*)(ws + (size_t)kN * kD * 4);                         //  6.29 MB
  __bf16* bbuf = (__bf16*)(ws + (size_t)kN * kD * 4 + (size_t)kN * kD * 2);   // 30.87 MB
  float*  thr  = (float*)(ws + (size_t)kN * kD * 4 + (size_t)kN * kD * 2 +
                          (size_t)kD * kLPAD * 2);                            // 16 KB

  linear_kernel <<<dim3(kD / 128, kN / 128), 256, 0, stream>>>(hidden, clfw, clfb, hbuf);
  colnorm_kernel<<<dim3((kLPAD + 255) / 256), 256, 0, stream>>>(weight, bbuf);
  rownorm_kernel<<<dim3(kN), 256, 0, stream>>>(hbuf, abuf);
  cos_gemm_kernel<<<dim3(kLPAD / 128, kN / 128), 256, 0, stream>>>(abuf, bbuf, out);
  topk_kernel   <<<dim3(kN), 256, 0, stream>>>(out, thr);
  finalize_kernel<<<dim3((int)(((size_t)kN * kL) / (256 * 4))), 256, 0, stream>>>(out, labels, thr);
}
